// Narayana_14542759264565
// MI455X (gfx1250) — compile-verified, matrix operand order checked by reading, not run on hardware
//
#include <hip/hip_runtime.h>
#include <cstdint>
#include <cstddef>

// ---------------------------------------------------------------- types ----
typedef __attribute__((ext_vector_type(16))) __bf16    v16bf;
typedef __attribute__((ext_vector_type(8)))  float     v8f;
typedef __attribute__((ext_vector_type(4)))  uint32_t  v4u;
typedef __attribute__((ext_vector_type(8)))  int       v8i;
typedef __attribute__((ext_vector_type(4)))  int       v4i;

union Frag {            // one WMMA 16-bit A/B operand: 16 bf16 per lane = 32B
    v16bf v;
    uint4 q[2];
};
union U128 {            // 16-byte staging chunk
    uint4    q;
    uint16_t h[8];
};

__device__ __forceinline__ uint16_t f2bf(float f) {
    uint32_t u = __builtin_bit_cast(uint32_t, f);
    uint32_t r = u + 0x7FFFu + ((u >> 16) & 1u);   // round-to-nearest-even
    return (uint16_t)(r >> 16);
}

__device__ __forceinline__ v8f wmma_bf16(const Frag& a, const Frag& b, v8f c) {
    return __builtin_amdgcn_wmma_f32_16x16x32_bf16(
        /*neg_a=*/false, a.v, /*neg_b=*/false, b.v,
        /*c_mod=*/(short)0, c, /*reuse_a=*/false, /*reuse_b=*/false);
}

// async global -> LDS copy, 16B per lane (512B per wave per instruction)
__device__ __forceinline__ void async_copy_b128(void* lds, const void* g) {
    uint32_t l = (uint32_t)(uintptr_t)lds;   // flat addr low 32 bits == LDS offset
    asm volatile("global_load_async_to_lds_b128 %0, %1, off"
                 :: "v"(l), "v"(g) : "memory");
}
__device__ __forceinline__ void wait_async0() {
    asm volatile("s_wait_asynccnt 0x0" ::: "memory");
}

// ------------------------------------------- Tensor Data Mover (TDM) -------
#if __has_builtin(__builtin_amdgcn_tensor_load_to_lds)
#define USE_TDM 1
#else
#define USE_TDM 0
#endif

#if USE_TDM
// DMA a 128-row x 32-col bf16 tile (row pitch = K elements) into LDS.
// D# per CDNA5 ISA ch.8: group0 = {count, lds_addr, global_addr, type=2},
// group1 = {data_size=2B, tensor/tile dims, dim0 stride}.
__device__ __forceinline__ void tdm_load_tile_128x32(void* lds, const void* g,
                                                     int K) {
    uint64_t ga = (uint64_t)(uintptr_t)g;
    v4u g0;
    g0[0] = 1u;                                           // count=1 (valid)
    g0[1] = (uint32_t)(uintptr_t)lds;                     // lds_addr (bytes)
    g0[2] = (uint32_t)ga;                                 // global_addr[31:0]
    g0[3] = (uint32_t)((ga >> 32) & 0x1FFFFFFu) | (2u << 30); // addr hi | type=2
    v8i g1;
    g1[0] = (int)(1u << 16);                              // data_size = 2 bytes
    g1[1] = (int)(32u << 16);                             // tensor_dim0 = 32
    g1[2] = (int)(128u << 16);                            // tensor_dim1 = 128
    g1[3] = (int)(32u << 16);                             // tile_dim0 = 32
    g1[4] = 128;                                          // tile_dim1 = 128
    g1[5] = K;                                            // tensor_dim0_stride
    g1[6] = 0;
    g1[7] = 0;
    v4i gz = {0, 0, 0, 0};
#if defined(__clang_major__) && __clang_major__ >= 23
    v8i z8 = {0, 0, 0, 0, 0, 0, 0, 0};
    __builtin_amdgcn_tensor_load_to_lds(g0, g1, gz, gz, z8, 0);
#else
    __builtin_amdgcn_tensor_load_to_lds(g0, g1, gz, gz, 0);
#endif
}
#endif

// ------------------------------------------------------- f32 -> bf16 -------
__global__ void f32_to_bf16_kernel(const float* __restrict__ in,
                                   uint16_t* __restrict__ out, size_t n) {
    size_t i = (size_t)blockIdx.x * blockDim.x + threadIdx.x;
    if (i < n) out[i] = f2bf(in[i]);
}

// --------------------------------------------- embedding + positional ------
__global__ void embed_pos_kernel(const int* __restrict__ tokens,
                                 const float* __restrict__ embed,
                                 float* __restrict__ x, int Tlen, int D) {
    int row = blockIdx.x;                 // 0 .. B*T-1
    int t   = row % Tlen;
    int tok = tokens[row];
    const float* er = embed + (size_t)tok * D;
    float* xr = x + (size_t)row * D;
    for (int c = threadIdx.x; c < D; c += blockDim.x) {
        int   i   = c >> 1;
        float div = __expf((float)(2 * i) * (-9.210340371976184f / (float)D));
        float ang = (float)t * div;
        float pos = (c & 1) ? __cosf(ang) : __sinf(ang);
        xr[c] = er[c] + pos;
    }
}

// ------------------------------------------------------------- layernorm ---
// one wave (32 lanes) per row of D=1024; output bf16
__global__ __launch_bounds__(256)
void layernorm_bf16_kernel(const float* __restrict__ x,
                           const float* __restrict__ w,
                           const float* __restrict__ b,
                           uint16_t* __restrict__ out, int D) {
    int wv = threadIdx.x >> 5, lane = threadIdx.x & 31;
    int row = blockIdx.x * 8 + wv;
    const float* xr = x + (size_t)row * D;
    float s = 0.f, s2 = 0.f;
    for (int c = lane; c < D; c += 32) { float v = xr[c]; s += v; s2 += v * v; }
    for (int m = 16; m; m >>= 1) {
        s  += __shfl_xor(s,  m, 32);
        s2 += __shfl_xor(s2, m, 32);
    }
    float mu   = s / (float)D;
    float var  = s2 / (float)D - mu * mu;
    float rstd = rsqrtf(var + 1e-5f);
    uint16_t* orow = out + (size_t)row * D;
    for (int c = lane; c < D; c += 32)
        orow[c] = f2bf((xr[c] - mu) * rstd * w[c] + b[c]);
}

// ------------------------------------------------------------------ GEMM ---
// C[M,N] = A[M,K](bf16) @ W[K,N](bf16) + bias, opt ReLU / residual,
// dual f32/bf16 output.
// 128x64 block tile, 8 waves x (32x32), K-step 32, double-buffered LDS.
// A tile staged by TDM (tensor_load_to_lds, TENSORcnt) when available, else
// per-lane async global->LDS (ASYNCcnt); W tile transposed via VGPRs.
__global__ __launch_bounds__(256)
void gemm_bf16_kernel(const uint16_t* __restrict__ A,
                      const uint16_t* __restrict__ W,
                      const float* __restrict__ bias,
                      const float* __restrict__ resid,
                      float* __restrict__ outF,
                      uint16_t* __restrict__ outB,
                      int M, int N, int K, int relu) {
    __shared__ uint16_t As[2][128][32];   // [buf][m][k]   16KB
    __shared__ uint16_t Bt[2][64][32];    // [buf][n][k]    8KB (transposed W)

    const int tid   = threadIdx.x;
    const int lane  = tid & 31;
    const int wv    = tid >> 5;
    const int lhalf = lane >> 4;
    const int l15   = lane & 15;
    const int bm    = blockIdx.y * 128;
    const int bn    = blockIdx.x * 64;
    const int mr    = (wv >> 1) * 32;     // 0,32,64,96
    const int nb    = (wv & 1) * 32;      // 0,32
    const int kp    = lhalf * 8;          // 16-bit operand K pattern base

    // W staging assignment: one k-row, 8 contiguous n per thread
    const int bk  = tid >> 3;             // 0..31
    const int bn0 = (tid & 7) * 8;        // 0..56

    v8f acc00 = {}, acc01 = {}, acc10 = {}, acc11 = {};

    // ---- prologue: stage k0 = 0 into buffer 0 ----
    {
#if USE_TDM
        if (wv == 0)
            tdm_load_tile_128x32(&As[0][0][0], A + (size_t)bm * K, K);
#else
#pragma unroll
        for (int i = 0; i < 4; ++i) {
            int c = tid + 256 * i;                 // 1024 chunks of 16B
            int row = c >> 2, col = (c & 3) * 8;
            async_copy_b128(&As[0][row][col], A + (size_t)(bm + row) * K + col);
        }
#endif
        U128 u;
        u.q = *(const uint4*)(W + (size_t)bk * N + bn + bn0);
#pragma unroll
        for (int i = 0; i < 8; ++i) Bt[0][bn0 + i][bk] = u.h[i];
#if USE_TDM
        __builtin_amdgcn_s_wait_tensorcnt(0);
#else
        wait_async0();
#endif
        __syncthreads();
    }

    for (int k0 = 0; k0 < K; k0 += 32) {
        const int  buf  = (k0 >> 5) & 1;
        const int  nxt  = buf ^ 1;
        const bool more = (k0 + 32) < K;

        // kick off next tile's copies before computing the current one
        U128 u = {};
        if (more) {
#if USE_TDM
            if (wv == 0)
                tdm_load_tile_128x32(&As[nxt][0][0],
                                     A + (size_t)bm * K + (k0 + 32), K);
#else
#pragma unroll
            for (int i = 0; i < 4; ++i) {
                int c = tid + 256 * i;
                int row = c >> 2, col = (c & 3) * 8;
                async_copy_b128(&As[nxt][row][col],
                                A + (size_t)(bm + row) * K + (k0 + 32) + col);
            }
#endif
            u.q = *(const uint4*)(W + (size_t)(k0 + 32 + bk) * N + bn + bn0);
            if (k0 + 64 < K)
                __builtin_prefetch(W + (size_t)(k0 + 64 + bk) * N + bn + bn0, 0, 1);
        }

        // compute on current buffer: 4 frags -> 4 WMMAs
        Frag a0, a1, b0, b1;
        a0.q[0] = *(const uint4*)(&As[buf][mr + l15][kp]);
        a0.q[1] = *(const uint4*)(&As[buf][mr + l15][kp + 16]);
        a1.q[0] = *(const uint4*)(&As[buf][mr + 16 + l15][kp]);
        a1.q[1] = *(const uint4*)(&As[buf][mr + 16 + l15][kp + 16]);
        b0.q[0] = *(const uint4*)(&Bt[buf][nb + l15][kp]);
        b0.q[1] = *(const uint4*)(&Bt[buf][nb + l15][kp + 16]);
        b1.q[0] = *(const uint4*)(&Bt[buf][nb + 16 + l15][kp]);
        b1.q[1] = *(const uint4*)(&Bt[buf][nb + 16 + l15][kp + 16]);
        acc00 = wmma_bf16(a0, b0, acc00);
        acc01 = wmma_bf16(a0, b1, acc01);
        acc10 = wmma_bf16(a1, b0, acc10);
        acc11 = wmma_bf16(a1, b1, acc11);

        // transpose-scatter next W tile into the other buffer
        if (more) {
#pragma unroll
            for (int i = 0; i < 8; ++i) Bt[nxt][bn0 + i][bk] = u.h[i];
        }
#if USE_TDM
        __builtin_amdgcn_s_wait_tensorcnt(0);
#else
        wait_async0();
#endif
        __syncthreads();
    }

    // epilogue: C layout -> VGPR r: M = r + 8*lhalf, N = lane&15
#pragma unroll
    for (int ti = 0; ti < 2; ++ti) {
        v8f aj0 = ti ? acc10 : acc00;
        v8f aj1 = ti ? acc11 : acc01;
#pragma unroll
        for (int r = 0; r < 8; ++r) {
            int m  = bm + mr + 16 * ti + r + lhalf * 8;
            int n0 = bn + nb + l15;
            size_t i0 = (size_t)m * N + n0;
            float v0 = aj0[r] + bias[n0];
            float v1 = aj1[r] + bias[n0 + 16];
            if (relu) { v0 = fmaxf(v0, 0.f); v1 = fmaxf(v1, 0.f); }
            if (resid) { v0 += resid[i0]; v1 += resid[i0 + 16]; }
            if (outF) { outF[i0] = v0; outF[i0 + 16] = v1; }
            if (outB) { outB[i0] = f2bf(v0); outB[i0 + 16] = f2bf(v1); }
        }
    }
}

// --------------------------------------------------------- flash attention -
// one wave per 16-query tile per (b,h); 32-key steps; causal; d_k=64
__global__ __launch_bounds__(256)
void flash_attn_kernel(const uint16_t* __restrict__ Q,
                       const uint16_t* __restrict__ Kb,
                       const uint16_t* __restrict__ Vb,
                       uint16_t* __restrict__ O,
                       int Hn, int Tn, int D) {
    __shared__ uint16_t Pbuf[8][16][32];   // P tile per wave
    __shared__ uint16_t Vt[8][64][32];     // V tile transposed per wave [chan][key]

    const int wv    = threadIdx.x >> 5;
    const int lane  = threadIdx.x & 31;
    const int lhalf = lane >> 4;
    const int l15   = lane & 15;
    const int ntile = Tn / 16;
    const int wid   = blockIdx.x * 8 + wv;
    const int mtile = wid % ntile;
    const int bh    = wid / ntile;
    const int b     = bh / Hn;
    const int h     = bh % Hn;
    const int m0    = mtile * 16;
    const size_t rowbase = (size_t)b * Tn;
    const int coff  = h * 64;
    const int kp    = lhalf * 8;

    // Q A-fragments (channels 0..31 and 32..63)
    Frag qa0, qa1;
    {
        const uint16_t* qr = Q + (rowbase + m0 + l15) * D + coff;
        qa0.q[0] = *(const uint4*)(qr + kp);
        qa0.q[1] = *(const uint4*)(qr + kp + 16);
        qa1.q[0] = *(const uint4*)(qr + 32 + kp);
        qa1.q[1] = *(const uint4*)(qr + 32 + kp + 16);
    }

    v8f o0 = {}, o1 = {}, o2 = {}, o3 = {};
    float mrow[8], lrow[8];
#pragma unroll
    for (int r = 0; r < 8; ++r) { mrow[r] = -3.0e38f; lrow[r] = 0.f; }

    const int jend = m0 + 16;
    for (int j = 0; j < jend; j += 32) {
        // K^T B-fragments: [chan-half][key-tile]
        Frag bk00, bk01, bk10, bk11;
        const uint16_t* kr0 = Kb + (rowbase + j + l15) * D + coff + kp;
        const uint16_t* kr1 = Kb + (rowbase + j + 16 + l15) * D + coff + kp;
        bk00.q[0] = *(const uint4*)(kr0);      bk00.q[1] = *(const uint4*)(kr0 + 16);
        bk10.q[0] = *(const uint4*)(kr0 + 32); bk10.q[1] = *(const uint4*)(kr0 + 48);
        bk01.q[0] = *(const uint4*)(kr1);      bk01.q[1] = *(const uint4*)(kr1 + 16);
        bk11.q[0] = *(const uint4*)(kr1 + 32); bk11.q[1] = *(const uint4*)(kr1 + 48);

        v8f s0 = {}, s1 = {};
        s0 = wmma_bf16(qa0, bk00, s0); s0 = wmma_bf16(qa1, bk10, s0);
        s1 = wmma_bf16(qa0, bk01, s1); s1 = wmma_bf16(qa1, bk11, s1);

        // stage V tile (32 keys x 64 chan) transposed into LDS
        {
            const uint16_t* vr = Vb + (rowbase + j + lane) * D + coff;
#pragma unroll
            for (int c8 = 0; c8 < 8; ++c8) {
                U128 u; u.q = *(const uint4*)(vr + c8 * 8);
#pragma unroll
                for (int i = 0; i < 8; ++i) Vt[wv][c8 * 8 + i][lane] = u.h[i];
            }
        }

        // online softmax over the 32 keys of this step
        const float scale = 0.125f;           // 1/sqrt(64)
        const int col0 = j + l15, col1 = j + 16 + l15;
        float p0[8], p1[8], tmax[8];
#pragma unroll
        for (int r = 0; r < 8; ++r) {
            int row = m0 + r + lhalf * 8;
            float a  = s0[r] * scale;
            float b2 = s1[r] * scale;
            if (col0 > row) a  = -3.0e38f;    // causal mask
            if (col1 > row) b2 = -3.0e38f;
            p0[r] = a; p1[r] = b2;
            float t = fmaxf(a, b2);
            t = fmaxf(t, __shfl_xor(t, 1, 32));
            t = fmaxf(t, __shfl_xor(t, 2, 32));
            t = fmaxf(t, __shfl_xor(t, 4, 32));
            t = fmaxf(t, __shfl_xor(t, 8, 32));
            tmax[r] = t;
        }
#pragma unroll
        for (int r = 0; r < 8; ++r) {
            float mn    = fmaxf(mrow[r], tmax[r]);
            float alpha = __expf(mrow[r] - mn);
            mrow[r] = mn;
            float e0 = __expf(p0[r] - mn);
            float e1 = __expf(p1[r] - mn);
            float rs = e0 + e1;
            rs += __shfl_xor(rs, 1, 32); rs += __shfl_xor(rs, 2, 32);
            rs += __shfl_xor(rs, 4, 32); rs += __shfl_xor(rs, 8, 32);
            lrow[r] = lrow[r] * alpha + rs;
            o0[r] *= alpha; o1[r] *= alpha; o2[r] *= alpha; o3[r] *= alpha;
            int row_l = r + lhalf * 8;
            Pbuf[wv][row_l][l15]      = f2bf(e0);
            Pbuf[wv][row_l][16 + l15] = f2bf(e1);
        }

        // DS ops are in-order within a wave; keep the compiler honest too
        asm volatile("s_wait_dscnt 0x0" ::: "memory");

        // P as A-fragment (16x32), V chunks as B-fragments (32x16)
        Frag pa, vb0, vb1, vb2, vb3;
        pa.q[0]  = *(const uint4*)(&Pbuf[wv][l15][kp]);
        pa.q[1]  = *(const uint4*)(&Pbuf[wv][l15][kp + 16]);
        vb0.q[0] = *(const uint4*)(&Vt[wv][l15][kp]);
        vb0.q[1] = *(const uint4*)(&Vt[wv][l15][kp + 16]);
        vb1.q[0] = *(const uint4*)(&Vt[wv][16 + l15][kp]);
        vb1.q[1] = *(const uint4*)(&Vt[wv][16 + l15][kp + 16]);
        vb2.q[0] = *(const uint4*)(&Vt[wv][32 + l15][kp]);
        vb2.q[1] = *(const uint4*)(&Vt[wv][32 + l15][kp + 16]);
        vb3.q[0] = *(const uint4*)(&Vt[wv][48 + l15][kp]);
        vb3.q[1] = *(const uint4*)(&Vt[wv][48 + l15][kp + 16]);
        o0 = wmma_bf16(pa, vb0, o0);
        o1 = wmma_bf16(pa, vb1, o1);
        o2 = wmma_bf16(pa, vb2, o2);
        o3 = wmma_bf16(pa, vb3, o3);
    }

    // normalize and write bf16 attention output
#pragma unroll
    for (int r = 0; r < 8; ++r) {
        size_t row = rowbase + m0 + r + lhalf * 8;
        float inv = 1.0f / lrow[r];
        uint16_t* orow = O + row * D + coff;
        orow[l15]      = f2bf(o0[r] * inv);
        orow[16 + l15] = f2bf(o1[r] * inv);
        orow[32 + l15] = f2bf(o2[r] * inv);
        orow[48 + l15] = f2bf(o3[r] * inv);
    }
}

// ----------------------------------------------------------------- host ----
static void launch_gemm(const uint16_t* A, const uint16_t* W, const float* bias,
                        const float* resid, float* outF, uint16_t* outB,
                        int M, int N, int K, int relu, hipStream_t stream) {
    dim3 grid(N / 64, M / 128);
    gemm_bf16_kernel<<<grid, 256, 0, stream>>>(A, W, bias, resid, outF, outB,
                                               M, N, K, relu);
}

extern "C" void kernel_launch(void* const* d_in, const int* in_sizes, int n_in,
                              void* d_out, int out_size, void* d_ws, size_t ws_size,
                              hipStream_t stream) {
    (void)in_sizes; (void)n_in; (void)out_size; (void)ws_size;
    const int L = 4, D = 1024, F = 4096, V = 32000, Bn = 2, Tn = 2048, H = 16;
    const int M = Bn * Tn;                           // 4096 token rows

    const int*   tokens = (const int*)  d_in[0];
    const float* embedW = (const float*)d_in[1];
    const float* Wq = (const float*)d_in[2];  const float* bq = (const float*)d_in[3];
    const float* Wk = (const float*)d_in[4];  const float* bk = (const float*)d_in[5];
    const float* Wv = (const float*)d_in[6];  const float* bv = (const float*)d_in[7];
    const float* Wo = (const float*)d_in[8];  const float* bo = (const float*)d_in[9];
    const float* ln1w = (const float*)d_in[10]; const float* ln1b = (const float*)d_in[11];
    const float* ff1w = (const float*)d_in[12]; const float* ff1b = (const float*)d_in[13];
    const float* ff2w = (const float*)d_in[14]; const float* ff2b = (const float*)d_in[15];
    const float* ln2w = (const float*)d_in[16]; const float* ln2b = (const float*)d_in[17];
    const float* projw = (const float*)d_in[18]; const float* projb = (const float*)d_in[19];
    float* out = (float*)d_out;

    // workspace carve-out
    char* p = (char*)d_ws;
    auto take = [&](size_t bytes) -> char* {
        char* r = p; p += (bytes + 255) & ~(size_t)255; return r;
    };
    uint16_t* wqb  = (uint16_t*)take((size_t)L * D * D * 2);
    uint16_t* wkb  = (uint16_t*)take((size_t)L * D * D * 2);
    uint16_t* wvb  = (uint16_t*)take((size_t)L * D * D * 2);
    uint16_t* wob  = (uint16_t*)take((size_t)L * D * D * 2);
    uint16_t* f1b  = (uint16_t*)take((size_t)L * D * F * 2);
    uint16_t* f2b  = (uint16_t*)take((size_t)L * F * D * 2);
    uint16_t* prb  = (uint16_t*)take((size_t)D * V * 2);
    float*    x    = (float*)   take((size_t)M * D * 4);
    uint16_t* xn   = (uint16_t*)take((size_t)M * D * 2);
    uint16_t* qb   = (uint16_t*)take((size_t)M * D * 2);
    uint16_t* kbuf = (uint16_t*)take((size_t)M * D * 2);
    uint16_t* vbuf = (uint16_t*)take((size_t)M * D * 2);
    uint16_t* ao   = (uint16_t*)take((size_t)M * D * 2);
    uint16_t* yn   = (uint16_t*)take((size_t)M * D * 2);
    uint16_t* h1   = (uint16_t*)take((size_t)M * F * 2);
    uint16_t* xb   = (uint16_t*)take((size_t)M * D * 2);

    auto conv = [&](const float* src, uint16_t* dst, size_t n) {
        f32_to_bf16_kernel<<<dim3((unsigned)((n + 255) / 256)), 256, 0, stream>>>(src, dst, n);
    };
    conv(Wq, wqb, (size_t)L * D * D);
    conv(Wk, wkb, (size_t)L * D * D);
    conv(Wv, wvb, (size_t)L * D * D);
    conv(Wo, wob, (size_t)L * D * D);
    conv(ff1w, f1b, (size_t)L * D * F);
    conv(ff2w, f2b, (size_t)L * F * D);
    conv(projw, prb, (size_t)D * V);

    // x = embed[tokens] + pos
    embed_pos_kernel<<<M, 256, 0, stream>>>(tokens, embedW, x, Tn, D);

    for (int l = 0; l < L; ++l) {
        const size_t od = (size_t)l * D * D, of1 = (size_t)l * D * F;
        // ln1 -> xn (bf16)
        layernorm_bf16_kernel<<<M / 8, 256, 0, stream>>>(x, ln1w + (size_t)l * D,
                                                         ln1b + (size_t)l * D, xn, D);
        // Q,K,V projections (bf16 out)
        launch_gemm(xn, wqb + od, bq + (size_t)l * D, nullptr, nullptr, qb,   M, D, D, 0, stream);
        launch_gemm(xn, wkb + od, bk + (size_t)l * D, nullptr, nullptr, kbuf, M, D, D, 0, stream);
        launch_gemm(xn, wvb + od, bv + (size_t)l * D, nullptr, nullptr, vbuf, M, D, D, 0, stream);
        // causal flash attention -> ao (bf16)
        flash_attn_kernel<<<(Bn * H * (Tn / 16)) / 8, 256, 0, stream>>>(qb, kbuf, vbuf, ao, H, Tn, D);
        // O-projection + residual -> x (f32)
        launch_gemm(ao, wob + od, bo + (size_t)l * D, x, x, nullptr, M, D, D, 0, stream);
        // ln2 -> yn (bf16)
        layernorm_bf16_kernel<<<M / 8, 256, 0, stream>>>(x, ln2w + (size_t)l * D,
                                                         ln2b + (size_t)l * D, yn, D);
        // FFN: relu(yn@W1+b1)@W2+b2 + residual
        launch_gemm(yn, f1b + of1, ff1b + (size_t)l * F, nullptr, nullptr, h1, M, F, D, 1, stream);
        launch_gemm(h1, f2b + of1, ff2b + (size_t)l * D, x, x, nullptr, M, D, F, 0, stream);
    }

    // final vocab projection: out = x @ proj_w + proj_b
    conv(x, xb, (size_t)M * D);
    launch_gemm(xb, prb, projb, nullptr, out, nullptr, M, V, D, 0, stream);
}
